// NTXentLoss_31988916421047
// MI455X (gfx1250) — matrix operand order repaired and checked by reading, not compiled, not run on hardware
//
#include <hip/hip_runtime.h>
#include <hip/hip_bf16.h>

typedef __attribute__((ext_vector_type(16))) _Float16 v16h;
typedef __attribute__((ext_vector_type(8)))  float    v8f;
typedef __attribute__((ext_vector_type(8)))  unsigned int v8u;

#define TT 4096   // sequence length (fixed by reference)
#define DD 64     // feature dim (fixed by reference; K=64 -> 2x WMMA K=32)
#define CT 4      // column tiles (of 16) per block -> 64 columns / block

// ---------------------------------------------------------------------------
// Kernel 1: per-batch combined masks + packed prefix scan -> compaction index
// ---------------------------------------------------------------------------
__global__ __launch_bounds__(1024)
void ntx_scan(const int* __restrict__ im1, const int* __restrict__ im2,
              const int* __restrict__ sh1, const int* __restrict__ sh2,
              int* __restrict__ ridxX, int* __restrict__ ridxY,
              int* __restrict__ counts) {
  const int b = blockIdx.x, tid = threadIdx.x;
  __shared__ int scan[1024];
  const size_t base = (size_t)b * TT + (size_t)tid * 4;
  int mx[4], my[4], sx = 0, sy = 0;
  #pragma unroll
  for (int i = 0; i < 4; ++i) {
    mx[i] = im1[base + i] * sh1[base + i];
    my[i] = im2[base + i] * sh2[base + i];
    sx += mx[i]; sy += my[i];
  }
  // pack x-count (lo16) and y-count (hi16); totals <= 4096 so no carry across
  scan[tid] = sx | (sy << 16);
  __syncthreads();
  for (int off = 1; off < 1024; off <<= 1) {
    int v = (tid >= off) ? scan[tid - off] : 0;
    __syncthreads();
    scan[tid] += v;
    __syncthreads();
  }
  const int incl = scan[tid];
  int ex = (incl & 0xFFFF) - sx;
  int ey = (incl >> 16) - sy;
  #pragma unroll
  for (int i = 0; i < 4; ++i) {
    ridxX[base + i] = mx[i] ? ex : -1; ex += mx[i];
    ridxY[base + i] = my[i] ? ey : -1; ey += my[i];
  }
  if (tid == 1023) { counts[2 * b] = incl & 0xFFFF; counts[2 * b + 1] = incl >> 16; }
}

// ---------------------------------------------------------------------------
// Kernel 2: normalize rows (one wave32 per row) + scatter to compacted f16
// ---------------------------------------------------------------------------
__global__ __launch_bounds__(256)
void ntx_normalize(const float* __restrict__ x, const float* __restrict__ y,
                   const int* __restrict__ ridxX, const int* __restrict__ ridxY,
                   unsigned* __restrict__ Xc, unsigned* __restrict__ Yc) {
  const int w = threadIdx.x >> 5, lane = threadIdx.x & 31;
  const size_t row = (size_t)blockIdx.x * 8 + w;      // global (b*T + t)
  const int b = (int)(row >> 12);                      // T == 4096
  const float2 vx = ((const float2*)(x + row * DD))[lane];
  const float2 vy = ((const float2*)(y + row * DD))[lane];
  float ssx = vx.x * vx.x + vx.y * vx.y;
  float ssy = vy.x * vy.x + vy.y * vy.y;
  #pragma unroll
  for (int o = 16; o > 0; o >>= 1) {                   // wave32 butterfly reduce
    ssx += __shfl_xor(ssx, o, 32);
    ssy += __shfl_xor(ssy, o, 32);
  }
  const float scx = 1.0f / fmaxf(sqrtf(ssx), 1e-12f);
  const float scy = 1.0f / fmaxf(sqrtf(ssy), 1e-12f);
  const int px = ridxX[row], py = ridxY[row];
  if (px >= 0) {
    _Float16 h0 = (_Float16)(vx.x * scx), h1 = (_Float16)(vx.y * scx);
    unsigned u = (unsigned)__builtin_bit_cast(unsigned short, h0) |
                 ((unsigned)__builtin_bit_cast(unsigned short, h1) << 16);
    Xc[((size_t)b * TT + px) * 32 + lane] = u;         // 32 dwords = 64 halves / row
  }
  if (py >= 0) {
    _Float16 h0 = (_Float16)(vy.x * scy), h1 = (_Float16)(vy.y * scy);
    unsigned u = (unsigned)__builtin_bit_cast(unsigned short, h0) |
                 ((unsigned)__builtin_bit_cast(unsigned short, h1) << 16);
    Yc[((size_t)b * TT + py) * 32 + lane] = u;
  }
}

// ---------------------------------------------------------------------------
// Kernel 3: zero pad rows up to the next multiple of 16
// ---------------------------------------------------------------------------
__global__ __launch_bounds__(256)
void ntx_pad(unsigned* __restrict__ Xc, unsigned* __restrict__ Yc,
             const int* __restrict__ counts) {
  const int b = blockIdx.x;
  const int nx = counts[2 * b], ny = counts[2 * b + 1];
  const int nxp = (nx + 15) & ~15, nyp = (ny + 15) & ~15;
  for (int i = threadIdx.x; i < (nxp - nx) * 32; i += blockDim.x)
    Xc[((size_t)b * TT + nx) * 32 + i] = 0u;
  for (int i = threadIdx.x; i < (nyp - ny) * 32; i += blockDim.x)
    Yc[((size_t)b * TT + ny) * 32 + i] = 0u;
}

// ---------------------------------------------------------------------------
// Kernel 4: WMMA similarity + streaming column sum-exp.
// One block = 64 columns (4 N-tiles) of one batch; 8 waves stride row tiles.
// Main loop is branch-free (full tiles); partial tail tile and diagonal
// extraction are peeled into epilogues.
// ---------------------------------------------------------------------------
__global__ __launch_bounds__(256)
void ntx_sim(const unsigned* __restrict__ Xc, const unsigned* __restrict__ Yc,
             const int* __restrict__ counts, float* __restrict__ partials) {
  const int b = blockIdx.y;
  const int col0 = blockIdx.x * (CT * 16);
  const int nx = counts[2 * b], ny = counts[2 * b + 1];
  const int m = nx < ny ? nx : ny;
  const int tid = threadIdx.x, w = tid >> 5, lane = tid & 31;
  float* part = partials + (size_t)b * gridDim.x + blockIdx.x;
  if (col0 >= m) { if (tid == 0) *part = 0.0f; return; }

  __shared__ unsigned ytile[CT * 16 * 32];   // 64 cols x 64 halves as dwords
  __shared__ float diag[CT * 16];
  __shared__ float wsum[8][CT][32];
  __shared__ float perv[CT * 16];

  const unsigned* Yb = Yc + ((size_t)b * TT + col0) * 32;
  for (int i = tid; i < CT * 16 * 32; i += 256) ytile[i] = Yb[i];
  __syncthreads();

  const int n = lane & 15, half = lane >> 4;

  // B-matrix (KxN, 16-bit) wave32 layout: lanes 0-15 K=0..7,16..23, lanes 16-31
  // K=8..15,24..31 (matches A). Y stored [col][k] so pairs are 1 dword.
  v16h blo[CT], bhi[CT];
  #pragma unroll
  for (int c = 0; c < CT; ++c) {
    v8u lo_u, hi_u;
    #pragma unroll
    for (int v = 0; v < 8; ++v) {
      const int kp = ((v >> 2) << 3) + (half << 2) + (v & 3);               // K/2
      lo_u[v] = ytile[(c * 16 + n) * 32 + kp];        // K in [0,32)
      hi_u[v] = ytile[(c * 16 + n) * 32 + 16 + kp];   // K in [32,64)
    }
    blo[c] = __builtin_bit_cast(v16h, lo_u);
    bhi[c] = __builtin_bit_cast(v16h, hi_u);
  }

  const unsigned* Xb = Xc + (size_t)b * TT * 32;
  const int ntfull = nx >> 4;                    // number of FULL row tiles
  float acc[CT] = {0.0f, 0.0f, 0.0f, 0.0f};

  // exp(sim/T) = exp2(dot * 10 * log2(e))
  const float EC = 14.426950408889634f;

  // ---- main loop: full tiles, branch-free ----
  for (int it = w; it < ntfull; it += 8) {
    // A-matrix (MxK, 16-bit) layout: M = lane&15; per-lane dwords form two
    // contiguous 16B runs -> compiler emits paired global_load_b128.
    const unsigned* xr = Xb + (size_t)(it * 16 + n) * 32;
    v8u alo_u, ahi_u;
    #pragma unroll
    for (int v = 0; v < 8; ++v) {
      const int kp = ((v >> 2) << 3) + (half << 2) + (v & 3);
      alo_u[v] = xr[kp];
      ahi_u[v] = xr[16 + kp];
    }
    const v16h alo = __builtin_bit_cast(v16h, alo_u);
    const v16h ahi = __builtin_bit_cast(v16h, ahi_u);
    #pragma unroll
    for (int c = 0; c < CT; ++c) {
      v8f d = {};
      d = __builtin_amdgcn_wmma_f32_16x16x32_f16(false, alo, false, blo[c], (short)0, d, false, false);
      d = __builtin_amdgcn_wmma_f32_16x16x32_f16(false, ahi, false, bhi[c], (short)0, d, false, false);
      float s = 0.0f;
      #pragma unroll
      for (int r = 0; r < 8; ++r)
        s += __builtin_amdgcn_exp2f(d[r] * EC);
      acc[c] += s;
    }
  }

  // ---- tail: single partial row tile, masked, one wave ----
  if ((nx & 15) && w == 7) {
    const int it = ntfull;
    const unsigned* xr = Xb + (size_t)(it * 16 + n) * 32;
    v8u alo_u, ahi_u;
    #pragma unroll
    for (int v = 0; v < 8; ++v) {
      const int kp = ((v >> 2) << 3) + (half << 2) + (v & 3);
      alo_u[v] = xr[kp];
      ahi_u[v] = xr[16 + kp];
    }
    const v16h alo = __builtin_bit_cast(v16h, alo_u);
    const v16h ahi = __builtin_bit_cast(v16h, ahi_u);
    const int rbase = it * 16 + 8 * half;        // C/D layout: M = r + 8*half
    #pragma unroll
    for (int c = 0; c < CT; ++c) {
      v8f d = {};
      d = __builtin_amdgcn_wmma_f32_16x16x32_f16(false, alo, false, blo[c], (short)0, d, false, false);
      d = __builtin_amdgcn_wmma_f32_16x16x32_f16(false, ahi, false, bhi[c], (short)0, d, false, false);
      float s = 0.0f;
      #pragma unroll
      for (int r = 0; r < 8; ++r)
        if (rbase + r < nx) s += __builtin_amdgcn_exp2f(d[r] * EC);
      acc[c] += s;
    }
  }

  // ---- diagonal epilogue: wave c recomputes the aligned tile for N-tile c ----
  #pragma unroll
  for (int c = 0; c < CT; ++c) {
    if (w == c) {                                // compile-time B-frag index
      const int it = (col0 >> 4) + c;
      const unsigned* xr = Xb + (size_t)(it * 16 + n) * 32;
      v8u alo_u, ahi_u;
      #pragma unroll
      for (int v = 0; v < 8; ++v) {
        const int kp = ((v >> 2) << 3) + (half << 2) + (v & 3);
        alo_u[v] = xr[kp];
        ahi_u[v] = xr[16 + kp];
      }
      const v16h alo = __builtin_bit_cast(v16h, alo_u);
      const v16h ahi = __builtin_bit_cast(v16h, ahi_u);
      v8f d = {};
      d = __builtin_amdgcn_wmma_f32_16x16x32_f16(false, alo, false, blo[c], (short)0, d, false, false);
      d = __builtin_amdgcn_wmma_f32_16x16x32_f16(false, ahi, false, bhi[c], (short)0, d, false, false);
      #pragma unroll
      for (int r = 0; r < 8; ++r)
        if ((r + 8 * half) == n)                 // M == N element
          diag[c * 16 + n] = d[r] * 10.0f;
    }
  }

  #pragma unroll
  for (int c = 0; c < CT; ++c) wsum[w][c][lane] = acc[c];
  __syncthreads();

  if (tid < CT * 16) {
    const int c = tid >> 4, j = tid & 15;
    float S = 0.0f;
    #pragma unroll
    for (int ww = 0; ww < 8; ++ww)
      S += wsum[ww][c][j] + wsum[ww][c][j + 16];
    const int jg = col0 + tid;
    perv[tid] = (jg < m)
        ? (__builtin_amdgcn_logf(S) * 0.6931471805599453f - diag[tid])
        : 0.0f;
  }
  __syncthreads();
  if (tid == 0) {
    float s = 0.0f;
    #pragma unroll
    for (int i = 0; i < CT * 16; ++i) s += perv[i];
    *part = s / (float)m;
  }
}

// ---------------------------------------------------------------------------
// Kernel 5: deterministic fixed-order reduction of partials -> scalar loss
// ---------------------------------------------------------------------------
__global__ __launch_bounds__(256)
void ntx_final(const float* __restrict__ partials, float* __restrict__ out,
               int n, float invB) {
  __shared__ float s[256];
  float a = 0.0f;
  for (int i = threadIdx.x; i < n; i += 256) a += partials[i];
  s[threadIdx.x] = a;
  __syncthreads();
  for (int off = 128; off > 0; off >>= 1) {
    if (threadIdx.x < off) s[threadIdx.x] += s[threadIdx.x + off];
    __syncthreads();
  }
  if (threadIdx.x == 0) out[0] = s[0] * invB;
}

// ---------------------------------------------------------------------------
extern "C" void kernel_launch(void* const* d_in, const int* in_sizes, int n_in,
                              void* d_out, int out_size, void* d_ws, size_t ws_size,
                              hipStream_t stream) {
  (void)n_in; (void)out_size; (void)ws_size;
  const float* x  = (const float*)d_in[0];
  const float* y  = (const float*)d_in[1];
  const int* im1  = (const int*)d_in[2];
  const int* im2  = (const int*)d_in[3];
  const int* sh1  = (const int*)d_in[4];
  const int* sh2  = (const int*)d_in[5];
  const int B = in_sizes[2] / TT;

  char* ws = (char*)d_ws;
  auto al = [](size_t v) { return (v + 255) & ~(size_t)255; };
  size_t o = 0;
  int* ridxX       = (int*)(ws + o);      o = al(o + (size_t)B * TT * 4);
  int* ridxY       = (int*)(ws + o);      o = al(o + (size_t)B * TT * 4);
  int* counts      = (int*)(ws + o);      o = al(o + (size_t)B * 8);
  unsigned* Xc     = (unsigned*)(ws + o); o = al(o + (size_t)B * TT * 32 * 4);
  unsigned* Yc     = (unsigned*)(ws + o); o = al(o + (size_t)B * TT * 32 * 4);
  float* partials  = (float*)(ws + o);    o = al(o + (size_t)B * (TT / (CT * 16)) * 4);

  hipLaunchKernelGGL(ntx_scan, dim3(B), dim3(1024), 0, stream,
                     im1, im2, sh1, sh2, ridxX, ridxY, counts);
  hipLaunchKernelGGL(ntx_normalize, dim3(B * TT / 8), dim3(256), 0, stream,
                     x, y, ridxX, ridxY, Xc, Yc);
  hipLaunchKernelGGL(ntx_pad, dim3(B), dim3(256), 0, stream, Xc, Yc, counts);
  hipLaunchKernelGGL(ntx_sim, dim3(TT / (CT * 16), B), dim3(256), 0, stream,
                     Xc, Yc, counts, partials);
  hipLaunchKernelGGL(ntx_final, dim3(1), dim3(256), 0, stream,
                     partials, (float*)d_out, B * (TT / (CT * 16)), 1.0f / (float)B);
}
